// EdgeGraphConvolutionalLayer_20023137534879
// MI455X (gfx1250) — matrix operand order repaired and checked by reading, not compiled, not run on hardware
//
#include <hip/hip_runtime.h>

#define B_  8
#define N_  1024
#define E_  32768
#define EF_ 4
#define D_  64

typedef __attribute__((ext_vector_type(16))) _Float16 v16h;
typedef __attribute__((ext_vector_type(8)))  _Float16 v8h;
typedef __attribute__((ext_vector_type(8)))  float    v8f;

// ---------------- workspace layout ----------------
// rd  : float    [B*EF*N]              @ 0          (128 KB)
// WhT : _Float16 [B*D*N]               @ 0x080000   (1 MB)
// Gm  : _Float16 [B*N*N]               @ 0x180000   (16 MB)
#define WS_RD_OFF  0
#define WS_WHT_OFF (512u * 1024u)
#define WS_GM_OFF  (1536u * 1024u)

// ---------------- 1. zero Gm (16 MB as float4 stores) ----------------
__global__ __launch_bounds__(256) void k_zero_gm(float4* __restrict__ p) {
    const int t = blockIdx.x * 256 + threadIdx.x;   // 1,048,576 threads
    p[t] = make_float4(0.f, 0.f, 0.f, 0.f);
}

// ---------------- 2. init degrees to 1.0 (self-loop) ----------------
__global__ __launch_bounds__(256) void k_init_rd(float* __restrict__ rd) {
    const int t = blockIdx.x * 256 + threadIdx.x;   // B*EF*N = 32768
    rd[t] = 1.0f;
}

// ---------------- 3. accumulate row degrees from edges ----------------
__global__ __launch_bounds__(256) void k_deg(const int* __restrict__ idx,
                                             const float* __restrict__ vals,
                                             float* __restrict__ rd) {
    const int t = blockIdx.x * 256 + threadIdx.x;   // B*E = 262144
    const int b = t >> 15;
    const int e = t & (E_ - 1);
    const int src = idx[(b * 2 + 0) * E_ + e];
    const float* v = vals + ((size_t)(b * E_ + e)) * EF_;
#pragma unroll
    for (int ef = 0; ef < EF_; ++ef)
        atomicAdd(&rd[(b * EF_ + ef) * N_ + src], v[ef]);
}

// ---------------- 4. rd = rsqrt(deg) ----------------
__global__ __launch_bounds__(256) void k_rsqrt(float* __restrict__ rd) {
    const int t = blockIdx.x * 256 + threadIdx.x;   // 32768
    rd[t] = rsqrtf(rd[t]);
}

// ---------------- 5. Wh -> WhT[b][d][n] in f16 ----------------
__global__ __launch_bounds__(256) void k_whT(const float* __restrict__ Wh,
                                             _Float16* __restrict__ WhT) {
    const int t = blockIdx.x * 256 + threadIdx.x;   // B*D*N = 524288
    const int n = t & (N_ - 1);
    const int d = (t >> 10) & (D_ - 1);
    const int b = t >> 16;
    WhT[t] = (_Float16)Wh[((size_t)(b * N_ + n)) * D_ + d];
}

// ---------------- 6. scatter folded edge weights into Gm ----------------
__global__ __launch_bounds__(256) void k_edges(const int* __restrict__ idx,
                                               const float* __restrict__ vals,
                                               const float* __restrict__ rd,
                                               _Float16* __restrict__ Gm) {
    const int t = blockIdx.x * 256 + threadIdx.x;   // B*E = 262144
    const int b = t >> 15;
    const int e = t & (E_ - 1);
    const int src = idx[(b * 2 + 0) * E_ + e];
    const int dst = idx[(b * 2 + 1) * E_ + e];
    const float* v = vals + ((size_t)(b * E_ + e)) * EF_;
    float s = 0.f;
#pragma unroll
    for (int ef = 0; ef < EF_; ++ef) {
        const float rs = rd[(b * EF_ + ef) * N_ + src];
        const float rt = rd[(b * EF_ + ef) * N_ + dst];
        s += v[ef] * rs * rt;
    }
    Gm[((size_t)(b * N_ + src)) * N_ + dst] = (_Float16)(0.25f * s);
}

// ---------------- 7. add self-loop diagonal ----------------
__global__ __launch_bounds__(256) void k_diag(const float* __restrict__ rd,
                                              _Float16* __restrict__ Gm) {
    const int t = blockIdx.x * 256 + threadIdx.x;   // B*N = 8192
    const int b = t >> 10;
    const int n = t & (N_ - 1);
    float s = 0.f;
#pragma unroll
    for (int ef = 0; ef < EF_; ++ef) {
        const float r = rd[(b * EF_ + ef) * N_ + n];
        s += r * r;
    }
    const size_t o = ((size_t)(b * N_ + n)) * N_ + n;
    Gm[o] = (_Float16)((float)Gm[o] + 0.25f * s);
}

// ---------------- 8. WMMA GEMM: out[b] = Gm[b] @ Wh[b] ----------------
static __device__ inline v16h pack16(v8h lo, v8h hi) {
    v16h r;
#pragma unroll
    for (int i = 0; i < 8; ++i) { r[i] = lo[i]; r[i + 8] = hi[i]; }
    return r;
}

__global__ __launch_bounds__(256) void k_gemm(const _Float16* __restrict__ Gm,
                                              const _Float16* __restrict__ WhT,
                                              float* __restrict__ out) {
    const int wave = threadIdx.x >> 5;
    const int lane = threadIdx.x & 31;
    const int half = lane >> 4;        // K-half selector per wave32 layout
    const int l16  = lane & 15;
    const int gw   = blockIdx.x * 8 + wave;   // 512 waves total
    const int b    = gw >> 6;
    const int nt   = gw & 63;
    const int n0   = nt * 16;

    // A fragment source: 16x32 A tile in this code's contiguous k order (B matches) — lane row = n0+l16,
// K runs 16*half .. 16*half+15 contiguous
    const _Float16* gmRow = Gm + ((size_t)(b * N_ + n0 + l16)) * N_;
    // B fragment source: column n = l16 of each 16-wide d tile,
    // K runs m+16*half .. m+16*half+15 contiguous in WhT row
    const _Float16* wt = WhT + (size_t)b * D_ * N_;
    const _Float16* wRow0 = wt + (size_t)( 0 + l16) * N_;
    const _Float16* wRow1 = wt + (size_t)(16 + l16) * N_;
    const _Float16* wRow2 = wt + (size_t)(32 + l16) * N_;
    const _Float16* wRow3 = wt + (size_t)(48 + l16) * N_;

    v8f c0 = {}, c1 = {}, c2 = {}, c3 = {};

    for (int m = 0; m < N_; m += 32) {
        // keep the next A strip in flight (gfx1250 global_prefetch path)
        __builtin_prefetch(gmRow + m + 128, 0, 1);

        const v8h a0 = *(const v8h*)(gmRow + m +     16 * half);
        const v8h a1 = *(const v8h*)(gmRow + m + 8 + 16 * half);
        const v16h A = pack16(a0, a1);

        const v16h Bt0 = *(const v16h*)(wRow0 + m + 16 * half);
        const v16h Bt1 = *(const v16h*)(wRow1 + m + 16 * half);
        const v16h Bt2 = *(const v16h*)(wRow2 + m + 16 * half);
        const v16h Bt3 = *(const v16h*)(wRow3 + m + 16 * half);

        c0 = __builtin_amdgcn_wmma_f32_16x16x32_f16(false, A, false, Bt0, (short)0, c0, false, false);
        c1 = __builtin_amdgcn_wmma_f32_16x16x32_f16(false, A, false, Bt1, (short)0, c1, false, false);
        c2 = __builtin_amdgcn_wmma_f32_16x16x32_f16(false, A, false, Bt2, (short)0, c2, false, false);
        c3 = __builtin_amdgcn_wmma_f32_16x16x32_f16(false, A, false, Bt3, (short)0, c3, false, false);
    }

    // C/D layout: VGPR j holds row M = j + 8*half, column n = l16
    float* orow = out + ((size_t)(b * N_ + n0)) * D_;
#pragma unroll
    for (int j = 0; j < 8; ++j) {
        const int M = j + 8 * half;
        orow[(size_t)M * D_ +  0 + l16] = c0[j];
        orow[(size_t)M * D_ + 16 + l16] = c1[j];
        orow[(size_t)M * D_ + 32 + l16] = c2[j];
        orow[(size_t)M * D_ + 48 + l16] = c3[j];
    }
}

extern "C" void kernel_launch(void* const* d_in, const int* in_sizes, int n_in,
                              void* d_out, int out_size, void* d_ws, size_t ws_size,
                              hipStream_t stream) {
    const float* Wh   = (const float*)d_in[0];
    const int*   idx  = (const int*)d_in[1];
    const float* vals = (const float*)d_in[2];
    float* out = (float*)d_out;

    char* ws = (char*)d_ws;
    float*     rd  = (float*)(ws + WS_RD_OFF);
    _Float16*  WhT = (_Float16*)(ws + WS_WHT_OFF);
    _Float16*  Gm  = (_Float16*)(ws + WS_GM_OFF);

    // 1) zero Gm: B*N*N*2 bytes / 16 = 1,048,576 float4 stores
    k_zero_gm<<<4096, 256, 0, stream>>>((float4*)Gm);
    // 2) deg = 1 (self loops)
    k_init_rd<<<(B_ * EF_ * N_) / 256, 256, 0, stream>>>(rd);
    // 3) deg += edge values (row sums)
    k_deg<<<(B_ * E_) / 256, 256, 0, stream>>>(idx, vals, rd);
    // 4) rd = rsqrt(deg)
    k_rsqrt<<<(B_ * EF_ * N_) / 256, 256, 0, stream>>>(rd);
    // 5) WhT = f16 transpose of Wh
    k_whT<<<(B_ * D_ * N_) / 256, 256, 0, stream>>>(Wh, WhT);
    // 6) scatter folded edges into Gm
    k_edges<<<(B_ * E_) / 256, 256, 0, stream>>>(idx, vals, rd, Gm);
    // 7) diagonal
    k_diag<<<(B_ * N_) / 256, 256, 0, stream>>>(rd, Gm);
    // 8) out = Gm @ Wh via v_wmma_f32_16x16x32_f16
    k_gemm<<<64, 256, 0, stream>>>(Gm, WhT, out);
}
